// CapNet_25305947308515
// MI455X (gfx1250) — matrix operand order repaired and checked by reading, not compiled, not run on hardware
//
#include <hip/hip_runtime.h>
#include <hip/hip_bf16.h>

// ---------------------------------------------------------------------------
// CapsNet forward for MI455X (gfx1250).
// Heavy convs run as mixed-precision implicit GEMMs on v_wmma_f32_16x16x32_f16
// (f16 in, f32 accumulate). Routing/decoder are fp32 VALU (trivial FLOPs).
// Workspace requirement: ~200 MB of d_ws.
// ---------------------------------------------------------------------------

typedef _Float16 half_t;
typedef __attribute__((ext_vector_type(16))) _Float16 v16h;
typedef __attribute__((ext_vector_type(8)))  _Float16 v8h;
typedef __attribute__((ext_vector_type(8)))  float    v8f;

#define BATCH 256

// ---------------------------------------------------------------------------
// A-fragment (16x32 f16, row M per lane): halves 0-15 = K kb..kb+15,
// kb = (lane<16)?0:16.  Two 16B loads.
// ---------------------------------------------------------------------------
__device__ __forceinline__ v16h load_frag_a(const half_t* __restrict__ row,
                                            int kc, int lane) {
    const int kb = (lane < 16) ? 0 : 16;
    const half_t* p = row + kc + kb;
    union { v16h v; v8h h[2]; } u;
    u.h[0] = *reinterpret_cast<const v8h*>(p);
    u.h[1] = *reinterpret_cast<const v8h*>(p + 8);
    return u.v;
}

// ---------------------------------------------------------------------------
// Weight / input packing kernels (fp32 -> f16, WMMA-friendly layouts)
// ---------------------------------------------------------------------------
__global__ void k_pack_w1(const float* __restrict__ w, half_t* __restrict__ w1p) {
    int tid = blockIdx.x * 256 + threadIdx.x;      // 256 oc * 96 k
    if (tid >= 256 * 96) return;
    int k = tid % 96, oc = tid / 96;
    w1p[tid] = (k < 81) ? (half_t)w[oc * 81 + k] : (half_t)0.f;
}

__global__ void k_im2col_x(const float* __restrict__ x, half_t* __restrict__ xcol) {
    int tid = blockIdx.x * 256 + threadIdx.x;      // N=102400 * K=96
    if (tid >= 102400 * 96) return;
    int k = tid % 96, n = tid / 96;
    half_t v = (half_t)0.f;
    if (k < 81) {
        int kh = k / 9, kw = k % 9;
        int b = n / 400, r = n % 400, oh = r / 20, ow = r % 20;
        v = (half_t)x[(b * 28 + oh + kh) * 28 + (ow + kw)];
    }
    xcol[tid] = v;
}

__global__ void k_pack_w2(const float* __restrict__ w, half_t* __restrict__ w2p) {
    int tid = blockIdx.x * 256 + threadIdx.x;      // [tap81][oc256][ic256]
    if (tid >= 81 * 256 * 256) return;
    int ic = tid % 256, r = tid / 256;
    int oc = r % 256, t = r / 256;
    w2p[tid] = (half_t)w[(oc * 256 + ic) * 81 + t];
}

// ---------------------------------------------------------------------------
// conv1 as GEMM: M=256 oc, N=102400 (b,20,20), K=96 (padded 81).
// Block = 4 waves, each wave owns a 16(M)x64(N) tile (4 accumulators).
// Output: h1 f16 NHWC  [b][y][x][oc]  (addr = n*256 + oc).
// ---------------------------------------------------------------------------
__global__ __launch_bounds__(128) void k_gemm_conv1(
        const half_t* __restrict__ w1p, const half_t* __restrict__ xcol,
        const float* __restrict__ bias, half_t* __restrict__ h1) {
    const int lane = threadIdx.x & 31;
    const int wave = threadIdx.x >> 5;
    const int n0 = blockIdx.x * 64;
    const int m0 = blockIdx.y * 64 + wave * 16;
    const int ml = lane & 15;
    const int kbB = (lane < 16) ? 0 : 16;

    v8f acc[4];
#pragma unroll
    for (int f = 0; f < 4; ++f) acc[f] = (v8f)(0.0f);

    const half_t* arow = w1p + (m0 + ml) * 96;
#pragma unroll
    for (int kc = 0; kc < 96; kc += 32) {
        v16h a = load_frag_a(arow, kc, lane);
#pragma unroll
        for (int f = 0; f < 4; ++f) {
            int n = n0 + f * 16 + ml;
            v16h b = *reinterpret_cast<const v16h*>(xcol + n * 96 + kc + kbB);
            acc[f] = __builtin_amdgcn_wmma_f32_16x16x32_f16(
                false, a, false, b, (short)0, acc[f], false, false);
        }
    }
    // C layout: lanes 0-15 hold M=r, lanes 16-31 M=r+8, N=lane&15.
    const int ocb = m0 + ((lane < 16) ? 0 : 8);
    float4 b0 = *reinterpret_cast<const float4*>(bias + ocb);
    float4 b1 = *reinterpret_cast<const float4*>(bias + ocb + 4);
    float bb[8] = {b0.x, b0.y, b0.z, b0.w, b1.x, b1.y, b1.z, b1.w};
#pragma unroll
    for (int f = 0; f < 4; ++f) {
        int n = n0 + f * 16 + ml;
        v8h o;
#pragma unroll
        for (int r = 0; r < 8; ++r) {
            float v = acc[f][r] + bb[r];
            o[r] = (half_t)(v > 0.f ? v : 0.f);           // ReLU
        }
        *reinterpret_cast<v8h*>(h1 + n * 256 + ocb) = o;  // 16B store
    }
}

// ---------------------------------------------------------------------------
// caps1 implicit GEMM: M=256 oc, N=9216 (b,6,6), K = 81 taps x 256 ic.
// Whole tap B tile (64n x 256ic f16 = 32KB) staged in LDS; 2 barriers and
// 32 WMMAs per tap (ic loop fully unrolled -> ds_load_b128/v_wmma pipeline).
// Output: h2 f32 [b][s=oh*6+ow][oc]  (addr = n*256 + oc), post-ReLU.
// ---------------------------------------------------------------------------
__global__ __launch_bounds__(128) void k_gemm_caps1(
        const half_t* __restrict__ w2p, const half_t* __restrict__ h1,
        const float* __restrict__ bias, float* __restrict__ h2) {
    __shared__ __align__(32) half_t bt[64 * 256];    // 32 KB of the 320 KB WGP LDS

    const int lane = threadIdx.x & 31;
    const int wave = threadIdx.x >> 5;
    const int n0 = blockIdx.x * 64;
    const int m0 = blockIdx.y * 64 + wave * 16;
    const int ml = lane & 15;
    const int kbB = (lane < 16) ? 0 : 16;

    // loader role: thread -> (n_local, 16-wide ic half)
    const int nl = threadIdx.x & 63;
    const int hs = threadIdx.x >> 6;                 // 0 or 1
    const int nt = n0 + nl;
    const int bimg = nt / 36, s = nt % 36, oh = s / 6, ow = s % 6;
    const half_t* bsrc_base = h1 + ((bimg * 20 + 2 * oh) * 20 + 2 * ow) * 256;

    v8f acc[4];
#pragma unroll
    for (int f = 0; f < 4; ++f) acc[f] = (v8f)(0.0f);

    const half_t* arow0 = w2p + (m0 + ml) * 256;

    for (int t = 0; t < 81; ++t) {
        const int kh = t / 9, kw = t % 9;
        const half_t* bsrc = bsrc_base + (kh * 20 + kw) * 256;
        const half_t* arow = arow0 + t * 256 * 256;
        __builtin_prefetch(arow + 256 * 256, 0, 1);                   // next tap A
        __builtin_prefetch(bsrc + ((kw == 8) ? 20 : 1) * 256, 0, 1);  // next tap B

        __syncthreads();                             // previous tap's compute done
#pragma unroll
        for (int c = 0; c < 8; ++c)                  // 8 x 32B per thread = 32KB
            *reinterpret_cast<v16h*>(&bt[nl * 256 + c * 32 + hs * 16]) =
                *reinterpret_cast<const v16h*>(bsrc + c * 32 + hs * 16);
        __syncthreads();                             // tile visible

#pragma unroll
        for (int kc = 0; kc < 256; kc += 32) {       // 8 chunks x 4 WMMA, no barriers
            v16h a = load_frag_a(arow, kc, lane);
#pragma unroll
            for (int f = 0; f < 4; ++f) {
                v16h b = *reinterpret_cast<const v16h*>(
                    &bt[(f * 16 + ml) * 256 + kc + kbB]);
                acc[f] = __builtin_amdgcn_wmma_f32_16x16x32_f16(
                    false, a, false, b, (short)0, acc[f], false, false);
            }
        }
    }
    const int ocb = m0 + ((lane < 16) ? 0 : 8);
    float4 b0 = *reinterpret_cast<const float4*>(bias + ocb);
    float4 b1 = *reinterpret_cast<const float4*>(bias + ocb + 4);
    float bb[8] = {b0.x, b0.y, b0.z, b0.w, b1.x, b1.y, b1.z, b1.w};
#pragma unroll
    for (int f = 0; f < 4; ++f) {
        int n = n0 + f * 16 + ml;
        float* op = h2 + n * 256 + ocb;
#pragma unroll
        for (int r = 0; r < 8; ++r) {
            float v = acc[f][r] + bb[r];
            op[r] = (v > 0.f ? v : 0.f);                      // ReLU
        }
    }
}

// ---------------------------------------------------------------------------
// u[b,i,j,d] = sum_e W'[i,j,d,e] * x_tile[b,i,j,e]   (faithful flat reshapes)
//   f = i*80 + j*8 + e ; channel c = f/360 ; spatial s = f%36
//   W'[d,e] = Wflat[(i*10+j)*128 + d*8 + e]
// Stored f16, d-contiguous: 32B rows for routing loads.
// ---------------------------------------------------------------------------
__global__ void k_compute_u(const float* __restrict__ h2,
                            const float* __restrict__ W,
                            half_t* __restrict__ u) {
    int tid = blockIdx.x * 256 + threadIdx.x;     // B*1152*10
    if (tid >= BATCH * 1152 * 10) return;
    int j = tid % 10, r = tid / 10;
    int i = r % 1152, b = r / 1152;
    float xt[8];
#pragma unroll
    for (int e = 0; e < 8; ++e) {
        int f = i * 80 + j * 8 + e;
        int c = f / 360;
        int s = f % 36;
        xt[e] = h2[(b * 36 + s) * 256 + c];
    }
    const float* wr = W + (i * 10 + j) * 128;
    v16h out;
#pragma unroll
    for (int d = 0; d < 16; ++d) {
        float a = 0.f;
#pragma unroll
        for (int e = 0; e < 8; ++e) a += wr[d * 8 + e] * xt[e];
        out[d] = (half_t)a;
    }
    *reinterpret_cast<v16h*>(u + (long)tid * 16) = out;
}

// ---------------------------------------------------------------------------
// One routing iteration. Block per image b; wave j (10 waves) owns class j.
//  iter 0: c = 0.1 exactly (softmax of zeros), b_ij written (no read).
//  iter 2: emit v_length -> d_out[0:2560] and masked -> workspace.
// ---------------------------------------------------------------------------
__global__ __launch_bounds__(320) void k_route(
        const half_t* __restrict__ u, const float* __restrict__ bias_r,
        float* __restrict__ b_ij, const float* __restrict__ y,
        float* __restrict__ vlen_out, float* __restrict__ masked, int iter) {
    __shared__ float red[10][32][16];
    __shared__ float sv[10][16];
    __shared__ float n2s[10];

    const int b = blockIdx.x;
    const int j = threadIdx.x >> 5;
    const int lane = threadIdx.x & 31;
    const int t = threadIdx.x;

    float acc[16];
#pragma unroll
    for (int d = 0; d < 16; ++d) acc[d] = 0.f;

    for (int i = lane; i < 1152; i += 32) {
        const long base = ((long)(b * 1152 + i)) * 10;
        float c;
        if (iter == 0) {
            c = 0.1f;
        } else {
            float lb[10], m = -1e30f;
#pragma unroll
            for (int k = 0; k < 10; ++k) { lb[k] = b_ij[base + k]; m = fmaxf(m, lb[k]); }
            float den = 0.f;
#pragma unroll
            for (int k = 0; k < 10; ++k) { lb[k] = expf(lb[k] - m); den += lb[k]; }
            c = lb[j] / den;
        }
        v16h ur = *reinterpret_cast<const v16h*>(u + (base + j) * 16);
#pragma unroll
        for (int d = 0; d < 16; ++d) acc[d] += c * (float)ur[d];
    }
#pragma unroll
    for (int d = 0; d < 16; ++d) red[j][lane][d] = acc[d];
    __syncthreads();

    if (t < 160) {                                   // s_j = sum + bias_r
        int jj = t >> 4, d = t & 15;
        float s = 0.f;
        for (int l = 0; l < 32; ++l) s += red[jj][l][d];
        sv[jj][d] = s + bias_r[jj * 16 + d];
    }
    __syncthreads();
    if (t < 10) {
        float s = 0.f;
        for (int d = 0; d < 16; ++d) { float x = sv[t][d]; s += x * x; }
        n2s[t] = s;
    }
    __syncthreads();
    if (t < 160) {                                   // squash
        int jj = t >> 4, d = t & 15;
        float n2 = n2s[jj];
        sv[jj][d] = sv[jj][d] * (n2 / (1.f + n2)) * rsqrtf(n2 + 1e-9f);
    }
    __syncthreads();

    if (iter < 2) {                                  // b_ij += u . v
        float vj[16];
#pragma unroll
        for (int d = 0; d < 16; ++d) vj[d] = sv[j][d];
        for (int i = lane; i < 1152; i += 32) {
            const long idx = ((long)(b * 1152 + i)) * 10 + j;
            v16h ur = *reinterpret_cast<const v16h*>(u + idx * 16);
            float a = 0.f;
#pragma unroll
            for (int d = 0; d < 16; ++d) a += (float)ur[d] * vj[d];
            b_ij[idx] = (iter == 0 ? 0.f : b_ij[idx]) + a;
        }
    } else {
        if (t < 10) {                                // v_length
            float nv = 0.f;
            for (int d = 0; d < 16; ++d) { float x = sv[t][d]; nv += x * x; }
            vlen_out[b * 10 + t] = sqrtf(nv + 1e-9f);
        }
        if (t < 16) {                                // masked: flat (B,16,10) @ y
            float a = 0.f;
#pragma unroll
            for (int k = 0; k < 10; ++k) {
                int fi = t * 10 + k;
                a += sv[fi >> 4][fi & 15] * y[b * 10 + k];
            }
            masked[b * 16 + t] = a;
        }
    }
}

// ---------------------------------------------------------------------------
// Dense fp32 FC: out[b,o] = act( in[b,:] . w[o,:] + bias[o] )   (w row-major)
// ---------------------------------------------------------------------------
__global__ void k_fc(const float* __restrict__ in, const float* __restrict__ w,
                     const float* __restrict__ bias, float* __restrict__ out,
                     int K, int N, int act) {
    int tid = blockIdx.x * 256 + threadIdx.x;
    if (tid >= BATCH * N) return;
    int o = tid % N, b = tid / N;
    const float* ir = in + b * K;
    const float* wr = w + o * K;
    float a = bias[o];
    for (int k = 0; k < K; k += 4)
        a += ir[k] * wr[k] + ir[k + 1] * wr[k + 1] +
             ir[k + 2] * wr[k + 2] + ir[k + 3] * wr[k + 3];
    if (act) a = 1.f / (1.f + expf(-a));
    out[tid] = a;
}

// ---------------------------------------------------------------------------
extern "C" void kernel_launch(void* const* d_in, const int* in_sizes, int n_in,
                              void* d_out, int out_size, void* d_ws, size_t ws_size,
                              hipStream_t stream) {
    const float* x       = (const float*)d_in[0];
    const float* y       = (const float*)d_in[1];
    const float* conv1_w = (const float*)d_in[2];
    const float* conv1_b = (const float*)d_in[3];
    const float* caps1_w = (const float*)d_in[4];
    const float* caps1_b = (const float*)d_in[5];
    const float* W       = (const float*)d_in[6];
    const float* bias_r  = (const float*)d_in[7];
    const float* fc1_w   = (const float*)d_in[8];
    const float* fc1_b   = (const float*)d_in[9];
    const float* fc2_w   = (const float*)d_in[10];
    const float* fc2_b   = (const float*)d_in[11];
    const float* fc3_w   = (const float*)d_in[12];
    const float* fc3_b   = (const float*)d_in[13];

    float* vlen  = (float*)d_out;                 // (B,10,1,1)
    float* recon = (float*)d_out + BATCH * 10;    // (B,784)

    // workspace carve (~200 MB), 256B aligned
    size_t off = 0;
    char* base = (char*)d_ws;
    auto carve = [&](size_t bytes) -> void* {
        void* p = base + off;
        off += (bytes + 255) & ~(size_t)255;
        return p;
    };
    half_t* w1p  = (half_t*)carve((size_t)256 * 96 * 2);
    half_t* xcol = (half_t*)carve((size_t)102400 * 96 * 2);
    half_t* h1   = (half_t*)carve((size_t)BATCH * 400 * 256 * 2);
    half_t* w2p  = (half_t*)carve((size_t)81 * 256 * 256 * 2);
    float*  h2   = (float*)carve((size_t)BATCH * 36 * 256 * 4);
    half_t* u    = (half_t*)carve((size_t)BATCH * 1152 * 10 * 16 * 2);
    float*  b_ij = (float*)carve((size_t)BATCH * 1152 * 10 * 4);
    float*  mskd = (float*)carve((size_t)BATCH * 16 * 4);
    float*  fc1o = (float*)carve((size_t)BATCH * 512 * 4);
    float*  fc2o = (float*)carve((size_t)BATCH * 1024 * 4);

    // conv1 (WMMA GEMM)
    k_pack_w1 <<<96, 256, 0, stream>>>(conv1_w, w1p);
    k_im2col_x<<<(102400 * 96) / 256, 256, 0, stream>>>(x, xcol);
    k_gemm_conv1<<<dim3(1600, 4), 128, 0, stream>>>(w1p, xcol, conv1_b, h1);

    // caps1 (WMMA implicit GEMM, full-tap LDS staging)
    k_pack_w2<<<(81 * 256 * 256) / 256, 256, 0, stream>>>(caps1_w, w2p);
    k_gemm_caps1<<<dim3(144, 4), 128, 0, stream>>>(w2p, h1, caps1_b, h2);

    // prediction vectors u
    k_compute_u<<<(BATCH * 1152 * 10) / 256, 256, 0, stream>>>(h2, W, u);

    // dynamic routing (3 iterations; last emits v_length + masked)
    for (int it = 0; it < 3; ++it)
        k_route<<<BATCH, 320, 0, stream>>>(u, bias_r, b_ij, y, vlen, mskd, it);

    // decoder
    k_fc<<<(BATCH * 512)  / 256, 256, 0, stream>>>(mskd, fc1_w, fc1_b, fc1o, 16,   512,  0);
    k_fc<<<(BATCH * 1024) / 256, 256, 0, stream>>>(fc1o, fc2_w, fc2_b, fc2o, 512,  1024, 0);
    k_fc<<<(BATCH * 784)  / 256, 256, 0, stream>>>(fc2o, fc3_w, fc3_b, recon, 1024, 784, 1);
}
